// LSTM_71571335020566
// MI455X (gfx1250) — matrix operand order repaired and checked by reading, not compiled, not run on hardware
//
#include <hip/hip_runtime.h>
#include <hip/hip_bf16.h>

typedef __attribute__((ext_vector_type(16))) _Float16 v16h;
typedef __attribute__((ext_vector_type(8)))  float    v8f;

#define HID 32
#define SEQT 2048

// ---- fast transcendentals: gfx1250 has hardware v_tanh_f32 ----
#if __has_builtin(__builtin_amdgcn_tanhf)
__device__ __forceinline__ float tanh_fast(float x) { return __builtin_amdgcn_tanhf(x); }
#elif __has_builtin(__builtin_amdgcn_exp2f)
__device__ __forceinline__ float tanh_fast(float x) {
  float e = __builtin_amdgcn_exp2f(-2.885390082f * x);
  return __builtin_amdgcn_rcpf(1.0f + e) * 2.0f - 1.0f;
}
#else
__device__ __forceinline__ float tanh_fast(float x) { return tanhf(x); }
#endif

// gates for i/f/o are pre-scaled by 0.5 in the weights, so
// sigmoid(raw) = 0.5*tanh(raw/2)+0.5 = fma(tanh(prescaled), 0.5, 0.5)
__device__ __forceinline__ float sig_from_half(float d) {
  return fmaf(tanh_fast(d), 0.5f, 0.5f);
}

__device__ __forceinline__ unsigned pack_h2(float a, float b) {
  unsigned short ua = __builtin_bit_cast(unsigned short, (_Float16)a);
  unsigned short ub = __builtin_bit_cast(unsigned short, (_Float16)b);
  return (unsigned)ua | ((unsigned)ub << 16);
}

union VU { unsigned u[8]; v16h v; };

#define WMMA(A, B, C) \
  __builtin_amdgcn_wmma_f32_16x16x32_f16(false, (A), false, (B), (short)0, (C), false, false)

__global__ __launch_bounds__(32) void lstm_fused_kernel(
    const float* __restrict__ X,     // [B, T]
    const float* __restrict__ w_ih,  // [128]
    const float* __restrict__ w_hh,  // [128, 32]
    const float* __restrict__ b_ih,  // [128]
    const float* __restrict__ b_hh,  // [128]
    const float* __restrict__ w_lin, // [32]
    const float* __restrict__ b_lin, // [1]
    float* __restrict__ out,         // [B]
    int Btot) {
  const int lane = threadIdx.x & 31;
  const int half = lane >> 4;   // 0: lanes 0-15, 1: lanes 16-31
  const int lr   = lane & 15;   // batch column within tile (N of D/B matrices)
  const int m0   = blockIdx.x * 16;
  if (m0 >= Btot) return;       // whole-wave exit only; EXEC all-ones below

  // Gate pre-scale: i,f,o rows scaled by 0.5 (sigmoid via tanh), g unscaled.
  // Gate n belongs to tile t = n/16: tiles 0-3 -> i,f ; 4,5 -> g ; 6,7 -> o.

  // ---- A tiles: w_hh rows (gate index) as f16, loaded once ----
  // A operand (16x32 f16), contiguous k order used here: lane(l%16)=row M; lanes0-15: K=0..7 (V0-3) & 8..15 (V4-7)
  //                       lanes16-31: K=16..23 & 24..31  ->  klo=16*half, khi=16*half+8
  v16h Aw[8];
  #pragma unroll
  for (int t = 0; t < 8; ++t) {
    const float gs = (t == 4 || t == 5) ? 1.0f : 0.5f;
    const float* wr = w_hh + (16 * t + lr) * HID;
    #pragma unroll
    for (int j = 0; j < 8; ++j) {
      Aw[t][j]     = (_Float16)(gs * wr[16 * half + j]);
      Aw[t][8 + j] = (_Float16)(gs * wr[16 * half + 8 + j]);
    }
  }

  // ---- per-lane f32 w_ih / bias for gate n = 16*t + r + 8*half (pre-scaled) ----
  float wihf[64], biasf[64];
  #pragma unroll
  for (int t = 0; t < 8; ++t) {
    const float gs = (t == 4 || t == 5) ? 1.0f : 0.5f;
    #pragma unroll
    for (int r = 0; r < 8; ++r) {
      int n = 16 * t + r + 8 * half;
      wihf[t * 8 + r]  = gs * w_ih[n];
      biasf[t * 8 + r] = gs * (b_ih[n] + b_hh[n]);
    }
  }

  // ---- state ----
  VU bu;                                  // B matrix = h^T (f16), starts at 0
  #pragma unroll
  for (int j = 0; j < 8; ++j) bu.u[j] = 0u;
  float cA[8], cB[8], hA[8], hB[8];       // cell/hidden for batch m=lr,
  #pragma unroll                          //   hdims A: 8*half+r, B: 16+8*half+r
  for (int r = 0; r < 8; ++r) { cA[r] = 0.f; cB[r] = 0.f; hA[r] = 0.f; hB[r] = 0.f; }

  const float* xp = X + (size_t)(m0 + lr) * (size_t)SEQT;

  for (int tt = 0; tt < SEQT; tt += 4) {
    const float4 x4 = *(const float4*)(xp + tt);   // 4 timesteps of this lane's row
    #pragma unroll
    for (int jj = 0; jj < 4; ++jj) {
      const float xv = (jj == 0) ? x4.x : (jj == 1) ? x4.y : (jj == 2) ? x4.z : x4.w;
      const v16h Bh = bu.v;
      v8f c0, c1;

      // ---- i gates (tiles 0,1) ----
      #pragma unroll
      for (int r = 0; r < 8; ++r) {
        c0[r] = fmaf(xv, wihf[0 * 8 + r], biasf[0 * 8 + r]);
        c1[r] = fmaf(xv, wihf[1 * 8 + r], biasf[1 * 8 + r]);
      }
      v8f d0 = WMMA(Aw[0], Bh, c0);
      v8f d1 = WMMA(Aw[1], Bh, c1);
      float iA[8], iB[8];
      #pragma unroll
      for (int r = 0; r < 8; ++r) { iA[r] = sig_from_half(d0[r]); iB[r] = sig_from_half(d1[r]); }

      // ---- f gates (tiles 2,3): c *= sigmoid(f) ----
      #pragma unroll
      for (int r = 0; r < 8; ++r) {
        c0[r] = fmaf(xv, wihf[2 * 8 + r], biasf[2 * 8 + r]);
        c1[r] = fmaf(xv, wihf[3 * 8 + r], biasf[3 * 8 + r]);
      }
      d0 = WMMA(Aw[2], Bh, c0);
      d1 = WMMA(Aw[3], Bh, c1);
      #pragma unroll
      for (int r = 0; r < 8; ++r) { cA[r] *= sig_from_half(d0[r]); cB[r] *= sig_from_half(d1[r]); }

      // ---- g gates (tiles 4,5, unscaled): c += i * tanh(g) ----
      #pragma unroll
      for (int r = 0; r < 8; ++r) {
        c0[r] = fmaf(xv, wihf[4 * 8 + r], biasf[4 * 8 + r]);
        c1[r] = fmaf(xv, wihf[5 * 8 + r], biasf[5 * 8 + r]);
      }
      d0 = WMMA(Aw[4], Bh, c0);
      d1 = WMMA(Aw[5], Bh, c1);
      #pragma unroll
      for (int r = 0; r < 8; ++r) {
        cA[r] = fmaf(iA[r], tanh_fast(d0[r]), cA[r]);
        cB[r] = fmaf(iB[r], tanh_fast(d1[r]), cB[r]);
      }

      // ---- o gates (tiles 6,7): h = sigmoid(o) * tanh(c) ----
      #pragma unroll
      for (int r = 0; r < 8; ++r) {
        c0[r] = fmaf(xv, wihf[6 * 8 + r], biasf[6 * 8 + r]);
        c1[r] = fmaf(xv, wihf[7 * 8 + r], biasf[7 * 8 + r]);
      }
      d0 = WMMA(Aw[6], Bh, c0);
      d1 = WMMA(Aw[7], Bh, c1);
      #pragma unroll
      for (int r = 0; r < 8; ++r) {
        hA[r] = sig_from_half(d0[r]) * tanh_fast(cA[r]);
        hB[r] = sig_from_half(d1[r]) * tanh_fast(cB[r]);
      }

      // ---- repack h (f32, D-layout) -> h^T (f16, B-layout) via xor-16 exchange ----
      // half0 lane needs hdims 0..15:  own pkA (0..7)   + partner pkA (8..15)
      // half1 lane needs hdims 16..31: partner pkB (16..23) + own pkB (24..31)
      #pragma unroll
      for (int j = 0; j < 4; ++j) {
        unsigned pA = pack_h2(hA[2 * j], hA[2 * j + 1]);
        unsigned pB = pack_h2(hB[2 * j], hB[2 * j + 1]);
        unsigned sA = (unsigned)__shfl_xor((int)pA, 16, 32);
        unsigned sB = (unsigned)__shfl_xor((int)pB, 16, 32);
        bu.u[j]     = half ? sB : pA;
        bu.u[4 + j] = half ? pB : sA;
      }
    }
  }

  // ---- epilogue: out[m] = leaky_relu(h . w_lin + b_lin) ----
  float part = 0.0f;
  #pragma unroll
  for (int r = 0; r < 8; ++r) {
    part = fmaf(hA[r], w_lin[8 * half + r], part);
    part = fmaf(hB[r], w_lin[16 + 8 * half + r], part);
  }
  part += __shfl_xor(part, 16, 32);
  if (half == 0) {
    float o = part + b_lin[0];
    out[m0 + lr] = (o >= 0.0f) ? o : 0.01f * o;
  }
}

extern "C" void kernel_launch(void* const* d_in, const int* in_sizes, int n_in,
                              void* d_out, int out_size, void* d_ws, size_t ws_size,
                              hipStream_t stream) {
  (void)n_in; (void)out_size; (void)d_ws; (void)ws_size;
  const float* X     = (const float*)d_in[0];
  const float* w_ih  = (const float*)d_in[1];
  const float* w_hh  = (const float*)d_in[2];
  const float* b_ih  = (const float*)d_in[3];
  const float* b_hh  = (const float*)d_in[4];
  const float* w_lin = (const float*)d_in[5];
  const float* b_lin = (const float*)d_in[6];
  float* out = (float*)d_out;

  const int Btot = in_sizes[0] / SEQT;          // 8192
  const int grid = (Btot + 15) / 16;            // one 32-thread wave per 16 rows
  lstm_fused_kernel<<<grid, 32, 0, stream>>>(X, w_ih, w_hh, b_ih, b_hh,
                                             w_lin, b_lin, out, Btot);
}
